// PerturbationModel_4243427688897
// MI455X (gfx1250) — matrix operand order repaired and checked by reading, not compiled, not run on hardware
//
#include <hip/hip_runtime.h>
#include <math.h>

// Problem dims
#define G_ 3
#define S_ 10
#define N_ 5000
#define K_ 20
#define O_ 1000
#define OPAD 1024
#define KPAD 32
#define EPS_ 1e-6f

// Tiling
#define WAVES 4
#define RTILES 4
#define ROWS_PER_WAVE (RTILES * 16)                 // 64
#define ROWS_PER_BLOCK (WAVES * ROWS_PER_WAVE)      // 256
#define BLOCKS_PER_GS ((N_ + ROWS_PER_BLOCK - 1) / ROWS_PER_BLOCK)  // 20

typedef _Float16 f16;
typedef __attribute__((ext_vector_type(16))) _Float16 v16h;
typedef __attribute__((ext_vector_type(8)))  float    v8f;

union AFrag { v16h v; f16 h[16]; };

// ---------------- precompute: log_mix (f16, zero-padded to KPAD x OPAD) ----
__global__ void pm_logmix_kernel(const float* __restrict__ otu,
                                 const float* __restrict__ cweight,
                                 const float* __restrict__ ccomm,
                                 f16* __restrict__ logmix) {
  int idx = blockIdx.x * blockDim.x + threadIdx.x;
  if (idx >= G_ * KPAD * OPAD) return;
  int g   = idx / (KPAD * OPAD);
  int rem = idx % (KPAD * OPAD);
  int k   = rem / OPAD;
  int o   = rem % OPAD;
  f16 val = (f16)0.0f;
  if (k < K_ && o < O_) {
    float cw  = cweight[g];
    float mix = otu[k * O_ + o] * (1.0f - cw) + cw * ccomm[g * O_ + o];
    val = (f16)logf(mix + EPS_);
  }
  logmix[idx] = val;
}

// ---------------- precompute: log(comm_dist + eps), [G*S][KPAD] ------------
__global__ void pm_logcomm_kernel(const float* __restrict__ comm,
                                  float* __restrict__ logcomm) {
  int idx = blockIdx.x * blockDim.x + threadIdx.x;
  if (idx >= G_ * S_ * KPAD) return;
  int gs = idx / KPAD;
  int k  = idx % KPAD;
  int g  = gs / S_, s = gs % S_;
  float v = 0.0f;
  if (k < K_) v = logf(comm[k * G_ * S_ + g * S_ + s] + EPS_);
  logcomm[idx] = v;
}

// ---------------- main: fused GEMM (WMMA f16->f32) + LSE reduction ---------
__global__ __launch_bounds__(WAVES * 32)
void pm_main_kernel(const float* __restrict__ counts,
                    const f16*   __restrict__ logmix,
                    const float* __restrict__ logcomm,
                    const float* __restrict__ gamma_p,
                    float*       __restrict__ out) {
  __shared__ float sc_lds[WAVES][16][KPAD + 1];   // +1 pad vs bank conflicts

  const int tid  = threadIdx.x;
  const int wave = tid >> 5;
  const int lane = tid & 31;
  const int hi   = lane >> 4;      // half-wave select
  const int lm   = lane & 15;

  const int gs  = blockIdx.x / BLOCKS_PER_GS;   // 0..29
  const int blk = blockIdx.x % BLOCKS_PER_GS;
  const int g   = gs / S_;

  const float gamma = gamma_p[0];
  const size_t cbase = (size_t)gs * N_ * O_;

  // B columns for this lane (this code's contiguous-k B order: col = lane&15,
  // halves 0..15 hold K-rows hi*16 .. hi*16+15 -> 32 contiguous bytes)
  const f16* bb    = logmix + (size_t)g * KPAD * OPAD;
  const f16* bcol0 = bb + (size_t)lm        * OPAD + hi * 16;
  const f16* bcol1 = bb + (size_t)(lm + 16) * OPAD + hi * 16;

  const int rowWave = blk * ROWS_PER_BLOCK + wave * ROWS_PER_WAVE;

  v8f acc[RTILES][2];
  #pragma unroll
  for (int r = 0; r < RTILES; ++r) { v8f z = {}; acc[r][0] = z; acc[r][1] = z; }

  // Per-row A pointers (ISA A layout: lane holds row = lane&15), clamped
  const float* aptr[RTILES];
  #pragma unroll
  for (int r = 0; r < RTILES; ++r) {
    int row = rowWave + r * 16 + lm;
    if (row > N_ - 1) row = N_ - 1;
    aptr[r] = counts + cbase + (size_t)row * O_;
  }

  // ---- full o-chunks: c = 0..30 (o = 0..991) ----
  for (int c = 0; c < 31; ++c) {
    const int ob = c * 32;
    v16h b0 = *(const v16h*)(bcol0 + ob);
    v16h b1 = *(const v16h*)(bcol1 + ob);
    #pragma unroll
    for (int r = 0; r < RTILES; ++r) {
      // A in the same contiguous order as B: lanes 0-15 take o-offsets {0..15},
      // lanes 16-31 need {16..31}
      const float* ap = aptr[r] + ob + hi * 16;
      float4 f0 = *(const float4*)(ap);
      float4 f1 = *(const float4*)(ap + 4);
      float4 f2 = *(const float4*)(ap + 8);
      float4 f3 = *(const float4*)(ap + 12);
      AFrag a;
      a.h[0]=(f16)f0.x;  a.h[1]=(f16)f0.y;  a.h[2]=(f16)f0.z;  a.h[3]=(f16)f0.w;
      a.h[4]=(f16)f1.x;  a.h[5]=(f16)f1.y;  a.h[6]=(f16)f1.z;  a.h[7]=(f16)f1.w;
      a.h[8]=(f16)f2.x;  a.h[9]=(f16)f2.y;  a.h[10]=(f16)f2.z; a.h[11]=(f16)f2.w;
      a.h[12]=(f16)f3.x; a.h[13]=(f16)f3.y; a.h[14]=(f16)f3.z; a.h[15]=(f16)f3.w;
      acc[r][0] = __builtin_amdgcn_wmma_f32_16x16x32_f16(
          false, a.v, false, b0, (short)0, acc[r][0], false, false);
      acc[r][1] = __builtin_amdgcn_wmma_f32_16x16x32_f16(
          false, a.v, false, b1, (short)0, acc[r][1], false, false);
    }
  }

  // ---- tail chunk c = 31: only o = 992..999 valid; B is 0 for o>=1000 ----
  {
    const int ob = 31 * 32;
    v16h b0 = *(const v16h*)(bcol0 + ob);
    v16h b1 = *(const v16h*)(bcol1 + ob);
    #pragma unroll
    for (int r = 0; r < RTILES; ++r) {
      AFrag a;
      #pragma unroll
      for (int i = 0; i < 16; ++i) a.h[i] = (f16)0.0f;
      if (hi == 0) {  // lanes 0-15 own o-offsets 0..7 of this chunk (992..999)
        const float* ap = aptr[r] + ob;
        float4 f0 = *(const float4*)(ap);
        float4 f1 = *(const float4*)(ap + 4);
        a.h[0]=(f16)f0.x; a.h[1]=(f16)f0.y; a.h[2]=(f16)f0.z; a.h[3]=(f16)f0.w;
        a.h[4]=(f16)f1.x; a.h[5]=(f16)f1.y; a.h[6]=(f16)f1.z; a.h[7]=(f16)f1.w;
      }
      acc[r][0] = __builtin_amdgcn_wmma_f32_16x16x32_f16(
          false, a.v, false, b0, (short)0, acc[r][0], false, false);
      acc[r][1] = __builtin_amdgcn_wmma_f32_16x16x32_f16(
          false, a.v, false, b1, (short)0, acc[r][1], false, false);
    }
  }

  // ---- epilogue: stage D tiles to LDS, per-row LSE over k=0..19 ----
  float partial = 0.0f;
  const float* lc = logcomm + gs * KPAD;
  for (int r = 0; r < RTILES; ++r) {
    __syncthreads();
    // D layout: lanes 0-15 -> M=j, lanes 16-31 -> M=8+j; col = lane&15
    #pragma unroll
    for (int j = 0; j < 8; ++j) {
      sc_lds[wave][hi * 8 + j][lm]      = acc[r][0][j];
      sc_lds[wave][hi * 8 + j][lm + 16] = acc[r][1][j];
    }
    __syncthreads();
    const int n = rowWave + r * 16 + lane;   // lanes 0..15 each own one row
    if (lane < 16 && n < N_) {
      float v[K_];
      float mx = -INFINITY;
      #pragma unroll
      for (int k = 0; k < K_; ++k) {
        v[k] = sc_lds[wave][lane][k] + lc[k];
        mx = fmaxf(mx, v[k]);
      }
      float sum = 0.0f;
      #pragma unroll
      for (int k = 0; k < K_; ++k) sum += expf(gamma * (v[k] - mx));
      partial += mx + logf(sum);
    }
  }

  // wave reduction + one atomic per wave
  #pragma unroll
  for (int off = 16; off > 0; off >>= 1)
    partial += __shfl_xor(partial, off, 32);
  if (lane == 0) atomicAdd(out, partial);
}

extern "C" void kernel_launch(void* const* d_in, const int* in_sizes, int n_in,
                              void* d_out, int out_size, void* d_ws, size_t ws_size,
                              hipStream_t stream) {
  (void)in_sizes; (void)n_in; (void)out_size; (void)ws_size;
  const float* counts  = (const float*)d_in[0];
  const float* otu     = (const float*)d_in[1];
  const float* comm    = (const float*)d_in[2];
  const float* cweight = (const float*)d_in[3];
  const float* ccomm   = (const float*)d_in[4];
  const float* gamma   = (const float*)d_in[5];
  float* out = (float*)d_out;

  f16*   logmix  = (f16*)d_ws;                                          // 192 KB
  float* logcomm = (float*)((char*)d_ws + (size_t)G_ * KPAD * OPAD * 2); // 3.75 KB

  hipMemsetAsync(d_out, 0, sizeof(float), stream);

  {
    int n = G_ * KPAD * OPAD;
    pm_logmix_kernel<<<(n + 255) / 256, 256, 0, stream>>>(otu, cweight, ccomm, logmix);
  }
  {
    int n = G_ * S_ * KPAD;
    pm_logcomm_kernel<<<(n + 255) / 256, 256, 0, stream>>>(comm, logcomm);
  }
  pm_main_kernel<<<G_ * S_ * BLOCKS_PER_GS, WAVES * 32, 0, stream>>>(
      counts, logmix, logcomm, gamma, out);
}
